// SCF_19000935317968
// MI455X (gfx1250) — matrix operand order repaired and checked by reading, not compiled, not run on hardware
//
#include <hip/hip_runtime.h>
#include <hip/hip_bf16.h>

#define NN    96
#define NP    9216        // 96*96
#define BATCH 32

typedef __attribute__((ext_vector_type(16))) __bf16 v16bf;
typedef __attribute__((ext_vector_type(8)))  float  v8f;

__device__ __forceinline__ float sigmoidf(float x) {
    return 1.0f / (1.0f + __expf(-x));
}

// ---------------------------------------------------------------------------
// G[g][i][h] = exp(-0.5 * torus_d2(i,h) * sigma_g^2), g = field*2 + {ex=0,in=1}
// ---------------------------------------------------------------------------
__global__ void build_g_kernel(float* __restrict__ G) {
    int idx = blockIdx.x * blockDim.x + threadIdx.x;
    if (idx >= 6 * NP) return;
    int g = idx / NP;
    int p = idx % NP;
    int i = p / NN, h = p % NN;
    float s2 = (g == 0) ? 3.5f * 3.5f
             : (g == 1) ? 6.3f * 6.3f
             : (g == 2) ? 5.3f * 5.3f
             : (g == 3) ? 11.8f * 11.8f
             : (g == 4) ? 3.5f * 3.5f
                        : 6.2f * 6.2f;
    float d = fabsf((float)(i - h));
    d = fminf(d, (float)NN - d);
    G[idx] = __expf(-0.5f * d * d * s2);
}

// ---------------------------------------------------------------------------
// T[g][h][j] = sum_k na_f[h][k] * G[g][j][k]   (na @ G^T, G symmetric)
// ---------------------------------------------------------------------------
__global__ void mmT_kernel(const float* __restrict__ G,
                           const float* __restrict__ na_v,
                           const float* __restrict__ na_a,
                           const float* __restrict__ na_m,
                           float* __restrict__ T) {
    int idx = blockIdx.x * blockDim.x + threadIdx.x;
    if (idx >= 6 * NP) return;
    int g = idx / NP;
    int f = g >> 1;
    int p = idx % NP;
    int h = p / NN, j = p % NN;
    const float* na = (f == 0) ? na_v : (f == 1) ? na_a : na_m;
    const float* Gr  = G + (size_t)g * NP + (size_t)j * NN;
    const float* nar = na + (size_t)h * NN;
    float s = 0.0f;
#pragma unroll 4
    for (int k = 0; k < NN; ++k) s += nar[k] * Gr[k];
    T[(size_t)g * NP + (size_t)h * NN + j] = s;
}

// ---------------------------------------------------------------------------
// NB[f][i][j] = l_ex * sum_h Gex[i][h]*Tex[h][j] - l_in * sum_h Gin[i][h]*Tin[h][j]
// ---------------------------------------------------------------------------
__global__ void nb_kernel(const float* __restrict__ G,
                          const float* __restrict__ T,
                          float* __restrict__ NB) {
    int idx = blockIdx.x * blockDim.x + threadIdx.x;
    if (idx >= 3 * NP) return;
    int f = idx / NP;
    int p = idx % NP;
    int i = p / NN, j = p % NN;
    float lex = (f == 0) ? 1.6f  : (f == 1) ? 1.0f : 3.8f;
    float lin = (f == 0) ? 1.23f : (f == 1) ? 0.8f : 3.3f;
    const float* Gex = G + (size_t)(2 * f)     * NP + (size_t)i * NN;
    const float* Gin = G + (size_t)(2 * f + 1) * NP + (size_t)i * NN;
    const float* Tex = T + (size_t)(2 * f)     * NP + j;
    const float* Tin = T + (size_t)(2 * f + 1) * NP + j;
    float se = 0.0f, si = 0.0f;
#pragma unroll 4
    for (int h = 0; h < NN; ++h) {
        se += Gex[h] * Tex[(size_t)h * NN];
        si += Gin[h] * Tin[(size_t)h * NN];
    }
    NB[idx] = lex * se - lin * si;
}

// ---------------------------------------------------------------------------
// Split-K GEMM: Cpart[kc][b][n] = sum_{k in chunk kc} A[b][k] * B[n][k]
//   A: 32 x K (video/audio), B: 9216 x K (rf), ksplit K-chunks.
// One wave per (n_tile, k_chunk); computes both 16-row M tiles (b 0..15 and
// 16..31) sharing the B fragment. fp32 loads -> in-register bf16 -> WMMA with
// fp32 accumulation. 576*ksplit waves keeps enough loads in flight to cover
// the HBM bandwidth-delay product at 23.3 TB/s.
// ---------------------------------------------------------------------------
__global__ void ext_gemm_kernel(const float* __restrict__ A,
                                const float* __restrict__ B,
                                float* __restrict__ Cpart,
                                int K, int n_tiles, int ksplit) {
    int wave = (int)((blockIdx.x * blockDim.x + threadIdx.x) >> 5);
    if (wave >= n_tiles * ksplit) return;      // wave-uniform: EXEC stays all-1s
    int nt = wave % n_tiles;
    int kc = wave / n_tiles;
    int klen = K / ksplit;
    int kbeg = kc * klen;
    int kend = kbeg + klen;

    int lane = threadIdx.x & 31;
    int half = lane >> 4;
    int l15  = lane & 15;
    int n_base = nt * 16;

    const float* Arow0 = A + (size_t)l15 * K;            // batch rows 0..15
    const float* Arow1 = A + (size_t)(16 + l15) * K;     // batch rows 16..31
    const float* Brow  = B + (size_t)(n_base + l15) * K; // rf row for column l15

    v8f acc0 = {};
    v8f acc1 = {};
#pragma unroll 2
    for (int k0 = kbeg; k0 < kend; k0 += 32) {
        int ka = k0 + half * 8;    // A: lanes<16 -> K 0..7 & 16..23; lanes>=16 -> +8
        int kb = k0 + half * 8;    // B: lanes<16 -> K 0..7 & 16..23; lanes>=16 -> +8
        v16bf a0, a1, bm;
#pragma unroll
        for (int t = 0; t < 8; ++t) {
            a0[t]     = (__bf16)Arow0[ka + t];
            a0[8 + t] = (__bf16)Arow0[ka + 16 + t];
            a1[t]     = (__bf16)Arow1[ka + t];
            a1[8 + t] = (__bf16)Arow1[ka + 16 + t];
        }
#pragma unroll
        for (int t = 0; t < 8; ++t) {
            bm[t]     = (__bf16)Brow[kb + t];
            bm[8 + t] = (__bf16)Brow[kb + 16 + t];
        }

        acc0 = __builtin_amdgcn_wmma_f32_16x16x32_bf16(
            false, a0, false, bm, (short)0, acc0, false, false);
        acc1 = __builtin_amdgcn_wmma_f32_16x16x32_bf16(
            false, a1, false, bm, (short)0, acc1, false, false);
    }
    // C/D layout: VGPR r -> M = r + 8*half, N = lane&15
    float* Cd = Cpart + (size_t)kc * BATCH * NP;
#pragma unroll
    for (int r = 0; r < 8; ++r) {
        int m = half * 8 + r;
        Cd[(size_t)m * NP + n_base + l15]        = acc0[r];
        Cd[(size_t)(16 + m) * NP + n_base + l15] = acc1[r];
    }
}

// ---------------------------------------------------------------------------
// Final elementwise update over (32, 96, 96); sums split-K partials in fixed
// order (deterministic, unlike float atomics).
// ---------------------------------------------------------------------------
__global__ void combine_kernel(const float* __restrict__ ext_v,
                               const float* __restrict__ ext_a,
                               const float* __restrict__ NB,
                               const float* __restrict__ na_v,
                               const float* __restrict__ na_a,
                               const float* __restrict__ na_m,
                               float* __restrict__ out, int ksplit) {
    int idx = blockIdx.x * blockDim.x + threadIdx.x;
    if (idx >= BATCH * NP) return;
    int ij = idx % NP;

    float ev = 0.0f, ea = 0.0f;
    for (int c = 0; c < ksplit; ++c) {
        ev += ext_v[(size_t)c * BATCH * NP + idx];
        ea += ext_a[(size_t)c * BATCH * NP + idx];
    }
    float nv  = NB[ij];
    float na_ = NB[NP + ij];
    float nm  = NB[2 * NP + ij];
    float av = na_v[ij], aa = na_a[ij], am = na_m[ij];

    const float c23 = 2.0f / 3.0f, c13 = 1.0f / 3.0f;
    float sv   = sigmoidf((ev + nv - 3.0f + 1.0f * am) * 0.3f);   // UPS_V=3, FV=1, P_V=0.3
    float navn = av * c23 + c13 * sv;
    float sa   = sigmoidf((ea + na_ - 3.0f + 1.0f * am) * 0.3f);  // UPS_A=3, FA=1, P_A=0.3
    float naan = aa * c23 + c13 * sa;
    float sm   = sigmoidf((7.0f * navn + 3.0f * naan + nm - 3.0f) * 0.3f); // K_V=7, K_A=3
    out[idx] = am * c23 + c13 * sm;
}

// ---------------------------------------------------------------------------
extern "C" void kernel_launch(void* const* d_in, const int* in_sizes, int n_in,
                              void* d_out, int out_size, void* d_ws, size_t ws_size,
                              hipStream_t stream) {
    const float* video = (const float*)d_in[0];   // 32 x 1024
    const float* audio = (const float*)d_in[1];   // 32 x 2048
    const float* rf_v  = (const float*)d_in[2];   // 9216 x 1024
    const float* rf_a  = (const float*)d_in[3];   // 9216 x 2048
    const float* na_v  = (const float*)d_in[4];   // 9216
    const float* na_a  = (const float*)d_in[5];   // 9216
    const float* na_m  = (const float*)d_in[6];   // 9216
    float* out = (float*)d_out;                   // 32 x 9216

    // Pick split-K factor by available workspace (deterministic either way).
    int ksplit = 4;
    {
        size_t need = ((size_t)15 * NP + (size_t)2 * ksplit * BATCH * NP) * sizeof(float);
        if (ws_size < need) ksplit = 1;
    }

    float* ws = (float*)d_ws;
    float* G     = ws;                       // 6 * 9216
    float* T     = G  + 6 * NP;              // 6 * 9216
    float* NB    = T  + 6 * NP;              // 3 * 9216
    float* ext_v = NB + 3 * NP;              // ksplit * 32 * 9216
    float* ext_a = ext_v + (size_t)ksplit * BATCH * NP;   // ksplit * 32 * 9216

    // small separable DoG pipeline (fp32 exact)
    build_g_kernel<<<(6 * NP + 255) / 256, 256, 0, stream>>>(G);
    mmT_kernel<<<(6 * NP + 255) / 256, 256, 0, stream>>>(G, na_v, na_a, na_m, T);
    nb_kernel<<<(3 * NP + 255) / 256, 256, 0, stream>>>(G, T, NB);

    // big memory-bound GEMMs on the WMMA path: 576 N-tiles x ksplit K-chunks
    const int n_tiles = NP / 16;                 // 576
    const int threads = 128;                     // 4 waves per block
    const int waves   = n_tiles * ksplit;
    const int blocks  = (waves * 32 + threads - 1) / threads;
    ext_gemm_kernel<<<blocks, threads, 0, stream>>>(video, rf_v, ext_v, 1024, n_tiles, ksplit);
    ext_gemm_kernel<<<blocks, threads, 0, stream>>>(audio, rf_a, ext_a, 2048, n_tiles, ksplit);

    combine_kernel<<<(BATCH * NP + 255) / 256, 256, 0, stream>>>(
        ext_v, ext_a, NB, na_v, na_a, na_m, out, ksplit);
}
